// Deformable_convolution_36438502540169
// MI455X (gfx1250) — matrix operand order repaired and checked by reading, not compiled, not run on hardware
//
#include <hip/hip_runtime.h>
#include <math.h>

// Problem constants (fixed by the reference)
#define B_    4
#define CIN   256
#define COUT  256
#define H_    64
#define W_    64
#define DG    2
#define KK    9
#define CG    128          // Cin / DG
#define HW    4096         // H*W
#define KD    2304         // Cin*9 == DG*CG*KK  (same K for both GEMMs)
#define NTOT  16384        // B*H*W
#define OMLD  64           // padded offset-conv output channels (54 -> 64)

typedef __attribute__((ext_vector_type(16))) __bf16 v16bf;
typedef __attribute__((ext_vector_type(8)))  float  v8f;

struct __align__(16) Q4 { unsigned x, y, z, w; };
union V16U { struct { Q4 lo; Q4 hi; } q; v16bf v; };

// fp32 -> bf16 round-to-nearest-even (pure bit math; no __bf16 scalar ops)
__device__ __forceinline__ unsigned short f2bf(float f) {
  unsigned u = __float_as_uint(f);
  u += 0x7FFFu + ((u >> 16) & 1u);
  return (unsigned short)(u >> 16);
}

// A-matrix 16x32 bf16 per-lane load: one contiguous 16-element chunk
__device__ __forceinline__ v16bf load_a16(const unsigned short* p) {
  V16U t;
  t.q.lo = *(const Q4*)(p);
  t.q.hi = *(const Q4*)(p + 8);
  return t.v;
}
// B-matrix 32x16 bf16 per-lane load: one contiguous 16-element chunk
__device__ __forceinline__ v16bf load_b16(const unsigned short* p) {
  V16U t;
  t.q.lo = *(const Q4*)(p);
  t.q.hi = *(const Q4*)(p + 8);
  return t.v;
}

// ---- kernel 0: convert weights to bf16 GEMM layouts --------------------------
// Wb  [256][KD]   kk = (d*9 + ki)*128 + c   (matches sampled-matrix layout)
// WOb [64 ][KD]   kk = cin*9 + ki           (matches im2col layout), rows >=54 zero
__global__ void k_cvt_w(const float* __restrict__ wgt, const float* __restrict__ woff,
                        unsigned short* __restrict__ Wb, unsigned short* __restrict__ WOb) {
  int tid = blockIdx.x * 256 + threadIdx.x;
  if (tid < COUT * KD) {
    int o  = tid / KD, kk = tid % KD;
    int d  = kk / (KK * CG);
    int rm = kk % (KK * CG);
    int ki = rm / CG, c = rm % CG;
    Wb[tid] = f2bf(wgt[(o * CIN + d * CG + c) * KK + ki]);
  }
  if (tid < OMLD * KD) {
    int co = tid / KD, kk = tid % KD;
    float v = (co < 54) ? woff[co * KD + kk] : 0.0f;
    WOb[tid] = f2bf(v);
  }
}

// ---- kernel 1: regular im2col of x (bf16), rows = pixel n, cols = kk ---------
__global__ void k_im2col(const float* __restrict__ x, unsigned short* __restrict__ X2) {
  int tid = blockIdx.x * 256 + threadIdx.x;     // CIN*NTOT threads
  int n   = tid & (NTOT - 1);                   // lanes -> consecutive pixels
  int cin = tid >> 14;
  int b = n >> 12, hw = n & (HW - 1);
  int h = hw >> 6, w = hw & 63;
  const float* xb = x + (((long)b * CIN + cin) << 12);
  unsigned short* dst = X2 + (long)n * KD + cin * KK;
#pragma unroll
  for (int ki = 0; ki < 9; ++ki) {
    int y = h + ki / 3 - 1, xx = w + ki % 3 - 1;
    float v = (y >= 0 && y < H_ && xx >= 0 && xx < W_) ? xb[(y << 6) + xx] : 0.0f;
    dst[ki] = f2bf(v);
  }
}

// ---- shared, software-pipelined WMMA core ------------------------------------
// One wave computes (MT*16)M x 64N. Two-phase ping-pong double buffering: the
// K-loop is unrolled x2 with named buffer sets, so loads write directly into
// the registers WMMA reads -- no loop-carried buffer copies are ever needed.
template<int MT>
__device__ __forceinline__ void wmma_core(const unsigned short* __restrict__ Arow,
                                          const unsigned short* __restrict__ Brow,
                                          v8f (&acc)[MT][4], int lane) {
  const unsigned short* ap = Arow + (long)(lane & 15) * KD + ((lane >> 4) << 4);
  const unsigned short* bp = Brow + (long)(lane & 15) * KD + ((lane >> 4) << 4);
  v16bf a0[MT], b0[4], a1[MT], b1[4];

  auto loadt = [&](v16bf (&a)[MT], v16bf (&b)[4], int k0) {
#pragma unroll
    for (int m = 0; m < MT; ++m) a[m] = load_a16(ap + (long)m * 16 * KD + k0);
#pragma unroll
    for (int j = 0; j < 4; ++j)  b[j] = load_b16(bp + (long)j * 16 * KD + k0);
  };
  auto comp = [&](v16bf (&a)[MT], v16bf (&b)[4]) {
#pragma unroll
    for (int j = 0; j < 4; ++j)
#pragma unroll
      for (int m = 0; m < MT; ++m)
        acc[m][j] = __builtin_amdgcn_wmma_f32_16x16x32_bf16(
            false, a[m], false, b[j], (short)0, acc[m][j], false, false);
  };

  loadt(a0, b0, 0);
  int k0 = 0;
  for (; k0 + 64 < KD; k0 += 64) {             // KD/32 is even: clean 2-step tail
    loadt(a1, b1, k0 + 32);
    __builtin_prefetch(ap + k0 + 2048, 0, 3);  // global_prefetch_b8 (L2 ahead)
    comp(a0, b0);
    loadt(a0, b0, k0 + 64);
    __builtin_prefetch(bp + k0 + 2048, 0, 3);
    comp(a1, b1);
  }
  loadt(a1, b1, k0 + 32);                       // k0 == KD - 64 here
  comp(a0, b0);
  comp(a1, b1);
}

// ---- kernel 2: offset/mask conv as WMMA GEMM: om[n][co] = WOb x X2 + b_off ---
__global__ void __launch_bounds__(128, 1)
k_gemm_om(const unsigned short* __restrict__ WOb,
          const unsigned short* __restrict__ X2,
          const float* __restrict__ b_off, float* __restrict__ om) {
  int lane = threadIdx.x & 31, wave = threadIdx.x >> 5;
  int o0 = wave * 16;                         // M = 64 covered by 4 waves
  long n0 = (long)blockIdx.x * 64;
  v8f acc[1][4] = {};
  wmma_core<1>(WOb + (long)o0 * KD, X2 + n0 * KD, acc, lane);
  int hi8 = (lane >> 4) << 3;
#pragma unroll
  for (int j = 0; j < 4; ++j) {
    long n = n0 + j * 16 + (lane & 15);
#pragma unroll
    for (int r = 0; r < 8; ++r) {
      int co = o0 + hi8 + r;
      if (co < 54) om[n * OMLD + co] = acc[0][j][r] + b_off[co];
    }
  }
}

// ---- kernel 3: deformable bilinear sampling -> bf16 B-matrix -----------------
// thread = (n, d, k); writes 128 contiguous bf16 as 16B-packed b128 stores
__global__ void k_sample(const float* __restrict__ x, const float* __restrict__ om,
                         unsigned short* __restrict__ S) {
  int tid = blockIdx.x * 256 + threadIdx.x;      // NTOT*DG*KK threads
  int n = tid / (DG * KK);
  int r = tid % (DG * KK);
  int d = r / KK, k = r % KK;
  int b = n >> 12, hw = n & (HW - 1), h = hw >> 6, w = hw & 63;
  const float* omr = om + (long)n * OMLD;
  float dy = omr[d * 18 + k];
  float dx = omr[d * 18 + 9 + k];
  float mz = omr[36 + d * 9 + k];
  float m  = 1.0f / (1.0f + __expf(-mz));        // sigmoid modulation
  float py = dy + (float)(h + k / 3 - 1);
  float px = dx + (float)(w + k % 3 - 1);
  float fy = floorf(py), fx = floorf(px);
  int y0 = (int)fy, x0 = (int)fx, y1 = y0 + 1, x1 = x0 + 1;
  float wy1 = py - fy, wx1 = px - fx, wy0 = 1.0f - wy1, wx0 = 1.0f - wx1;
  bool vy0 = (y0 >= 0) & (y0 < H_), vy1 = (y1 >= 0) & (y1 < H_);
  bool vx0 = (x0 >= 0) & (x0 < W_), vx1 = (x1 >= 0) & (x1 < W_);
  float w00 = (vy0 & vx0) ? wy0 * wx0 * m : 0.0f;
  float w01 = (vy0 & vx1) ? wy0 * wx1 * m : 0.0f;
  float w10 = (vy1 & vx0) ? wy1 * wx0 * m : 0.0f;
  float w11 = (vy1 & vx1) ? wy1 * wx1 * m : 0.0f;
  int cy0 = min(max(y0, 0), H_ - 1), cy1 = min(max(y1, 0), H_ - 1);
  int cx0 = min(max(x0, 0), W_ - 1), cx1 = min(max(x1, 0), W_ - 1);
  int i00 = (cy0 << 6) + cx0, i01 = (cy0 << 6) + cx1;
  int i10 = (cy1 << 6) + cx0, i11 = (cy1 << 6) + cx1;
  const float* xb = x + (((long)b * CIN + d * CG) << 12);
  unsigned short* dst = S + (long)n * KD + (d * KK + k) * CG;
  for (int c0 = 0; c0 < CG; c0 += 8) {
    unsigned pk[4];
#pragma unroll
    for (int i = 0; i < 4; ++i) {
      const float* xc0 = xb + ((long)(c0 + 2 * i) << 12);
      const float* xc1 = xb + ((long)(c0 + 2 * i + 1) << 12);
      float v0 = w00 * xc0[i00] + w01 * xc0[i01] + w10 * xc0[i10] + w11 * xc0[i11];
      float v1 = w00 * xc1[i00] + w01 * xc1[i01] + w10 * xc1[i10] + w11 * xc1[i11];
      pk[i] = (unsigned)f2bf(v0) | ((unsigned)f2bf(v1) << 16);
    }
    Q4 q; q.x = pk[0]; q.y = pk[1]; q.z = pk[2]; q.w = pk[3];
    *(Q4*)(dst + c0) = q;                        // one b128 store per 8 channels
  }
}

// ---- kernel 4: main WMMA GEMM: out(NCHW) = Wb x S + bias ---------------------
// wave tile 32M x 64N; block = 4 waves along M -> 128M x 64N per block
__global__ void __launch_bounds__(128, 1)
k_gemm_main(const unsigned short* __restrict__ Wb,
            const unsigned short* __restrict__ S,
            const float* __restrict__ bias, float* __restrict__ out) {
  int lane = threadIdx.x & 31, wave = threadIdx.x >> 5;
  int o0 = blockIdx.y * 128 + wave * 32;
  long n0 = (long)blockIdx.x * 64;
  v8f acc[2][4] = {};
  wmma_core<2>(Wb + (long)o0 * KD, S + n0 * KD, acc, lane);
  int hi8 = (lane >> 4) << 3;
#pragma unroll
  for (int m = 0; m < 2; ++m)
#pragma unroll
    for (int j = 0; j < 4; ++j) {
      long n = n0 + j * 16 + (lane & 15);
      int b = (int)(n >> 12), hw = (int)(n & (HW - 1));
#pragma unroll
      for (int r = 0; r < 8; ++r) {
        int o = o0 + m * 16 + hi8 + r;
        out[((long)(b * COUT + o) << 12) + hw] = acc[m][j][r] + bias[o];
      }
    }
}

extern "C" void kernel_launch(void* const* d_in, const int* in_sizes, int n_in,
                              void* d_out, int out_size, void* d_ws, size_t ws_size,
                              hipStream_t stream) {
  const float* x     = (const float*)d_in[0];
  const float* w_off = (const float*)d_in[1];
  const float* b_off = (const float*)d_in[2];
  const float* wgt   = (const float*)d_in[3];
  const float* bias  = (const float*)d_in[4];
  float* out = (float*)d_out;

  // Workspace layout (S aliases X2: strictly sequential uses)
  char* ws = (char*)d_ws;
  unsigned short* S   = (unsigned short*)(ws);               // 16384*2304*2 = 75,497,472 B
  float*          om  = (float*)(ws + 75497472);             // 16384*64*4   =  4,194,304 B
  unsigned short* Wb  = (unsigned short*)(ws + 79691776);    // 256*2304*2   =  1,179,648 B
  unsigned short* WOb = (unsigned short*)(ws + 80871424);    // 64*2304*2    =    294,912 B

  k_cvt_w    <<<2304, 256, 0, stream>>>(wgt, w_off, Wb, WOb);
  k_im2col   <<<16384, 256, 0, stream>>>(x, S);              // S used as X2 here
  k_gemm_om  <<<256, 128, 0, stream>>>(WOb, S, b_off, om);
  k_sample   <<<1152, 256, 0, stream>>>(x, om, S);           // overwrites all of S
  k_gemm_main<<<dim3(256, 2), 128, 0, stream>>>(Wb, S, bias, out);
}
